// GCN_22170621182138
// MI455X (gfx1250) — matrix operand order repaired and checked by reading, not compiled, not run on hardware
//
#include <hip/hip_runtime.h>
#include <hip/hip_bf16.h>

// ---------------------------------------------------------------------------
// GraphSAGE-pool (3 layers + BN + classifier) for gfx1250.
// bf16 activations, v_wmma_f32_16x16x32_bf16 GEMMs, TDM weight staging to LDS,
// segment-max via u32 atomic max on non-negative fp32 (post-ReLU values).
// ---------------------------------------------------------------------------

typedef __attribute__((ext_vector_type(16))) __bf16   v16bf;
typedef __attribute__((ext_vector_type(8)))  __bf16   v8bf;
typedef __attribute__((ext_vector_type(8)))  float    v8f;
typedef __attribute__((ext_vector_type(4)))  unsigned v4u;
typedef __attribute__((ext_vector_type(8)))  int      v8i;
typedef __attribute__((ext_vector_type(4)))  int      v4i;

union ABfrag { v16bf v; struct { v8bf lo, hi; } p; };

#ifndef __has_builtin
#define __has_builtin(x) 0
#endif
#if __has_builtin(__builtin_amdgcn_tensor_load_to_lds)
#define USE_TDM 1
#else
#define USE_TDM 0
#endif

#if USE_TDM
// 1-D TDM tile: nbytes (multiple of 8) from global -> LDS.
// D# per CDNA5 ISA 8.3/8.4: data_size=3 (8B units), tensor_dim0 = tile_dim0,
// unit stride; groups 2/3 zero (<=2D tensor).
__device__ __forceinline__ void tdm_load_1d(const void* gsrc, unsigned lds_off,
                                            unsigned nbytes)
{
    const unsigned n8 = nbytes >> 3;                      // elements of 8B
    const unsigned long long ga = (unsigned long long)gsrc;
    v4u g0;
    g0[0] = 1u;                                           // count=1, user mode
    g0[1] = lds_off;                                      // lds_addr
    g0[2] = (unsigned)(ga & 0xffffffffu);                 // global_addr[31:0]
    g0[3] = (unsigned)((ga >> 32) & 0x01ffffffu)          // global_addr[56:32]
          | (2u << 30);                                   // type=2 ("image")
    v8i g1;
    g1[0] = (int)(3u << 16);                              // wg_mask=0, data_size=8B
    g1[1] = (int)((n8 & 0xffffu) << 16);                  // tensor_dim0[15:0]
    g1[2] = (int)(((n8 >> 16) & 0xffffu) | (1u << 16));   // dim0[31:16], tensor_dim1=1
    g1[3] = (int)((n8 & 0xffffu) << 16);                  // tile_dim0
    g1[4] = 0;                                            // tile_dim1/2 unused
    g1[5] = (int)n8;                                      // tensor_dim0_stride[31:0]
    g1[6] = 0;
    g1[7] = 0;
    v4i z = {0, 0, 0, 0};
#if __clang_major__ >= 23
    v8i z8 = {0, 0, 0, 0, 0, 0, 0, 0};
    __builtin_amdgcn_tensor_load_to_lds(g0, g1, z, z, z8, 0);
#else
    __builtin_amdgcn_tensor_load_to_lds(g0, g1, z, z, 0);
#endif
}
#endif

// ---------------------------------------------------------------------------
// Fused (dual-)GEMM:  out = act( A1@W1 [+ A2@W2] + bias )
// A1: [nrows x 128] bf16 row-major. A2: bf16 or f32 (a2_f32) row-major,
// converted to bf16 fragments in-register. Wt: [128 cols][128 k] bf16
// pre-transposed (k-contiguous) so B fragments are two 16B LDS reads.
// act: 0=relu, 1=leaky(0.01), 2=none.  out_f32 may alias A2 (in-place is safe:
// each wave reads and writes only its own 16 rows, reads precede writes).
// ---------------------------------------------------------------------------
__global__ void __launch_bounds__(256)
sage_gemm(const __bf16* __restrict__ A1, const __bf16* __restrict__ Wt1,
          const void* A2, const __bf16* __restrict__ Wt2, int a2_f32,
          const float* __restrict__ bias,
          __bf16* __restrict__ out_bf, float* out_f32,
          int nrows, int act)
{
    __shared__ __align__(16) __bf16 ldsW[2 * 128 * 128];   // 64 KB of 320 KB WGP LDS
#if USE_TDM
    if (threadIdx.x < 32) {                                // wave 0 drives the TDM
        tdm_load_1d(Wt1, (unsigned)(unsigned long long)(void*)&ldsW[0], 32768u);
        if (A2)
            tdm_load_1d(Wt2, (unsigned)(unsigned long long)(void*)&ldsW[128 * 128], 32768u);
        __builtin_amdgcn_s_wait_tensorcnt(0);
    }
#else
    {
        const float4* s1 = (const float4*)Wt1;
        float4* d1 = (float4*)ldsW;
        for (int i = threadIdx.x; i < 2048; i += 256) d1[i] = s1[i];
        if (A2) {
            const float4* s2 = (const float4*)Wt2;
            float4* d2 = (float4*)(ldsW + 128 * 128);
            for (int i = threadIdx.x; i < 2048; i += 256) d2[i] = s2[i];
        }
    }
#endif
    __syncthreads();

    const int lane = threadIdx.x & 31;
    const int wave = threadIdx.x >> 5;
    const int tile = blockIdx.x * 8 + wave;        // 16-row tile per wave
    if (tile * 16 >= nrows) return;                // wave-uniform: EXEC stays all-1s

    const int half = lane >> 4;                    // ISA 16-bit A layout: K halves
    const int l16  = lane & 15;
    const int row  = tile * 16 + l16;

    v8f acc[8] = {};                               // 8 x (16x16 f32) tiles = 128 cols

    const __bf16* a1 = A1 + (size_t)row * 128;

    for (int k0 = 0; k0 < 128; k0 += 32) {
        ABfrag af;
        af.p.lo = *(const v8bf*)(a1 + k0 + half * 8);        // K = k0+0..7 / 8..15
        af.p.hi = *(const v8bf*)(a1 + k0 + 16 + half * 8);   // K = k0+16..23 / 24..31
        #pragma unroll
        for (int nt = 0; nt < 8; nt++) {
            ABfrag bf;
            const __bf16* w = ldsW + (size_t)(nt * 16 + l16) * 128 + k0 + half * 8;
            bf.p.lo = *(const v8bf*)(w);
            bf.p.hi = *(const v8bf*)(w + 16);
            acc[nt] = __builtin_amdgcn_wmma_f32_16x16x32_bf16(
                false, af.v, false, bf.v, (short)0, acc[nt], false, false);
        }
        if (A2) {
            ABfrag af2;
            if (a2_f32) {
                const float* a2f = (const float*)A2 + (size_t)row * 128 + k0 + half * 8;
                #pragma unroll
                for (int j = 0; j < 8; j++) af2.v[j]     = (__bf16)a2f[j];
                #pragma unroll
                for (int j = 0; j < 8; j++) af2.v[8 + j] = (__bf16)a2f[16 + j];
            } else {
                const __bf16* a2 = (const __bf16*)A2 + (size_t)row * 128;
                af2.p.lo = *(const v8bf*)(a2 + k0 + half * 8);
                af2.p.hi = *(const v8bf*)(a2 + k0 + 16 + half * 8);
            }
            #pragma unroll
            for (int nt = 0; nt < 8; nt++) {
                ABfrag bf;
                const __bf16* w = ldsW + 128 * 128 +
                                  (size_t)(nt * 16 + l16) * 128 + k0 + half * 8;
                bf.p.lo = *(const v8bf*)(w);
                bf.p.hi = *(const v8bf*)(w + 16);
                acc[nt] = __builtin_amdgcn_wmma_f32_16x16x32_bf16(
                    false, af2.v, false, bf.v, (short)0, acc[nt], false, false);
            }
        }
    }

    #pragma unroll
    for (int nt = 0; nt < 8; nt++) {
        const int col = nt * 16 + l16;             // C/D layout: N = lane%16
        const float bv = bias[col];
        #pragma unroll
        for (int r = 0; r < 8; r++) {              // M = r + 8*(lane/16)
            float v = acc[nt][r] + bv;
            if (act == 0)      v = v > 0.f ? v : 0.f;
            else if (act == 1) v = v > 0.f ? v : 0.01f * v;
            const size_t o = (size_t)(tile * 16 + half * 8 + r) * 128 + col;
            if (out_bf)  out_bf[o]  = (__bf16)v;
            if (out_f32) out_f32[o] = v;
        }
    }
}

// out[N x 16] = A @ WcT + bc  (single 16-col tile per wave)
__global__ void __launch_bounds__(256)
sage_classifier(const __bf16* __restrict__ A, const __bf16* __restrict__ WtC,
                const float* __restrict__ bc, float* __restrict__ out, int nrows)
{
    __shared__ __align__(16) __bf16 ldsW[16 * 128];
#if USE_TDM
    if (threadIdx.x < 32) {
        tdm_load_1d(WtC, (unsigned)(unsigned long long)(void*)&ldsW[0], 4096u);
        __builtin_amdgcn_s_wait_tensorcnt(0);
    }
#else
    {
        const float4* s = (const float4*)WtC;
        float4* d = (float4*)ldsW;
        for (int i = threadIdx.x; i < 256; i += 256) d[i] = s[i];
    }
#endif
    __syncthreads();
    const int lane = threadIdx.x & 31;
    const int wave = threadIdx.x >> 5;
    const int tile = blockIdx.x * 8 + wave;
    if (tile * 16 >= nrows) return;
    const int half = lane >> 4, l16 = lane & 15;
    const __bf16* a = A + (size_t)(tile * 16 + l16) * 128;
    v8f acc = {};
    for (int k0 = 0; k0 < 128; k0 += 32) {
        ABfrag af, bf;
        af.p.lo = *(const v8bf*)(a + k0 + half * 8);
        af.p.hi = *(const v8bf*)(a + k0 + 16 + half * 8);
        const __bf16* w = ldsW + (size_t)l16 * 128 + k0 + half * 8;
        bf.p.lo = *(const v8bf*)(w);
        bf.p.hi = *(const v8bf*)(w + 16);
        acc = __builtin_amdgcn_wmma_f32_16x16x32_bf16(
            false, af.v, false, bf.v, (short)0, acc, false, false);
    }
    const float b = bc[l16];
    #pragma unroll
    for (int r = 0; r < 8; r++)
        out[(size_t)(tile * 16 + half * 8 + r) * 16 + l16] = acc[r] + b;
}

// segment-max: one wave per edge, 4 feats/lane. hp >= 0 (post-ReLU) so
// u32 atomic max on the f32 bit pattern == float max; zero-init buffer also
// implements the deg>0 masking (isolated nodes stay exactly 0).
__global__ void __launch_bounds__(256)
sage_scatter_max(const __bf16* __restrict__ hp, const int* __restrict__ src,
                 const int* __restrict__ dst, unsigned* __restrict__ agg, int nedges)
{
    const int e = (blockIdx.x * 256 + threadIdx.x) >> 5;
    const int lane = threadIdx.x & 31;
    if (e >= nedges) return;
    const int s = src[e], d = dst[e];
    const __bf16* h = hp + (size_t)s * 128 + lane * 4;
    unsigned* a = agg + (size_t)d * 128 + lane * 4;
    #pragma unroll
    for (int i = 0; i < 4; i++) {
        float v = (float)h[i];
        atomicMax(a + i, __float_as_uint(v));
    }
}

__global__ void __launch_bounds__(256)
zero_u32(unsigned* __restrict__ p, int n4)
{
    int i = blockIdx.x * 256 + threadIdx.x;
    if (i < n4) ((uint4*)p)[i] = make_uint4(0u, 0u, 0u, 0u);
}

__global__ void __launch_bounds__(256)
cvt_f32_bf16(const float* __restrict__ in, __bf16* __restrict__ out, int n)
{
    int i = (blockIdx.x * 256 + threadIdx.x) * 4;
    if (i >= n) return;
    float4 v = *(const float4*)(in + i);
    out[i + 0] = (__bf16)v.x; out[i + 1] = (__bf16)v.y;
    out[i + 2] = (__bf16)v.z; out[i + 3] = (__bf16)v.w;
}

// W [K x Ncols] f32 row-major  ->  Wt [Ncols][K] bf16 (k-contiguous per col)
__global__ void __launch_bounds__(256)
wt_transpose(const float* __restrict__ W, __bf16* __restrict__ Wt, int K, int Ncols)
{
    int tid = blockIdx.x * 256 + threadIdx.x;
    if (tid >= K * Ncols) return;
    int n = tid / K, k = tid - n * K;
    Wt[(size_t)n * K + k] = (__bf16)W[(size_t)k * Ncols + n];
}

__global__ void __launch_bounds__(128)
bn_reduce(const float* __restrict__ x, float* __restrict__ sums,
          float* __restrict__ sumsq, int nrows)
{
    const int c = threadIdx.x;                 // 128 cols, coalesced row reads
    float s = 0.f, s2 = 0.f;
    for (int r = blockIdx.x; r < nrows; r += gridDim.x) {
        float v = x[(size_t)r * 128 + c];
        s += v; s2 += v * v;
    }
    atomicAdd(sums + c, s);
    atomicAdd(sumsq + c, s2);
}

__global__ void __launch_bounds__(128)
bn_finalize(const float* sums, const float* sumsq, const float* gamma,
            const float* beta, float* scale, float* shift, int nrows)
{
    int c = threadIdx.x;
    float inv = 1.0f / (float)nrows;
    float mu  = sums[c] * inv;
    float var = sumsq[c] * inv - mu * mu;
    float sc  = gamma[c] * rsqrtf(var + 1e-5f);
    scale[c] = sc;
    shift[c] = beta[c] - mu * sc;
}

__global__ void __launch_bounds__(256)
bn_apply(const float* __restrict__ x, const float* __restrict__ scale,
         const float* __restrict__ shift, __bf16* __restrict__ out, int n)
{
    int i = blockIdx.x * 256 + threadIdx.x;
    if (i >= n) return;
    int c = i & 127;
    float v = x[i] * scale[c] + shift[c];
    v = v > 0.f ? v : 0.01f * v;
    out[i] = (__bf16)v;
}

// ---------------------------------------------------------------------------
extern "C" void kernel_launch(void* const* d_in, const int* in_sizes, int n_in,
                              void* d_out, int out_size, void* d_ws, size_t ws_size,
                              hipStream_t stream)
{
    const int N = in_sizes[0] / 128;
    const int E = in_sizes[1];

    const float* node_feat = (const float*)d_in[0];
    const int*   src       = (const int*)d_in[1];
    const int*   dst       = (const int*)d_in[2];
    const float* wp[3]  = {(const float*)d_in[3],  (const float*)d_in[8],  (const float*)d_in[13]};
    const float* bp[3]  = {(const float*)d_in[4],  (const float*)d_in[9],  (const float*)d_in[14]};
    const float* wsm[3] = {(const float*)d_in[5],  (const float*)d_in[10], (const float*)d_in[15]};
    const float* wn[3]  = {(const float*)d_in[6],  (const float*)d_in[11], (const float*)d_in[16]};
    const float* bb[3]  = {(const float*)d_in[7],  (const float*)d_in[12], (const float*)d_in[17]};
    const float* gamma  = (const float*)d_in[18];
    const float* beta   = (const float*)d_in[19];
    const float* wc     = (const float*)d_in[20];
    const float* bc     = (const float*)d_in[21];

    // workspace carve: H0,H1 (bf16) + G32 (f32 agg / pre-BN) + bf16 weights + stats
    char* ws = (char*)d_ws;
    const size_t SZ_BF  = (size_t)N * 128 * 2;
    const size_t SZ_F32 = (size_t)N * 128 * 4;
    __bf16* H[2] = {(__bf16*)ws, (__bf16*)(ws + SZ_BF)};
    float*  G32  = (float*)(ws + 2 * SZ_BF);
    char*  wbase = ws + 2 * SZ_BF + SZ_F32;
    __bf16 *wpT[3], *wsT[3], *wnT[3];
    for (int i = 0; i < 3; i++) {
        wpT[i] = (__bf16*)(wbase + (size_t)(0 + i) * 32768);
        wsT[i] = (__bf16*)(wbase + (size_t)(3 + i) * 32768);
        wnT[i] = (__bf16*)(wbase + (size_t)(6 + i) * 32768);
    }
    __bf16* wcT  = (__bf16*)(wbase + 9 * 32768);
    float* stats = (float*)(wbase + 9 * 32768 + 4096);
    float *sums = stats, *sumsq = stats + 128, *scale = stats + 256, *shift = stats + 384;

    const int gemm_blocks  = (N / 16 + 7) / 8;
    const int ewave_blocks = (E * 32 + 255) / 256;
    const int elem4_blocks = (N * 128 / 4 + 255) / 256;
    const int elem_blocks  = (N * 128 + 255) / 256;

    for (int i = 0; i < 3; i++) {
        wt_transpose<<<64, 256, 0, stream>>>(wp[i],  wpT[i], 128, 128);
        wt_transpose<<<64, 256, 0, stream>>>(wsm[i], wsT[i], 128, 128);
        wt_transpose<<<64, 256, 0, stream>>>(wn[i],  wnT[i], 128, 128);
    }
    wt_transpose<<<8, 256, 0, stream>>>(wc, wcT, 128, 16);
    cvt_f32_bf16<<<elem4_blocks, 256, 0, stream>>>(node_feat, H[0], N * 128);

    int cur = 0;
    for (int layer = 0; layer < 3; layer++) {
        // hp = relu(h @ wp + bp)
        sage_gemm<<<gemm_blocks, 256, 0, stream>>>(
            H[cur], wpT[layer], (const void*)nullptr, (const __bf16*)nullptr, 0,
            bp[layer], H[cur ^ 1], (float*)nullptr, N, 0);
        // agg = segment_max(hp[src], dst) with zero identity
        zero_u32<<<elem4_blocks, 256, 0, stream>>>((unsigned*)G32, N * 128 / 4);
        sage_scatter_max<<<ewave_blocks, 256, 0, stream>>>(H[cur ^ 1], src, dst,
                                                           (unsigned*)G32, E);
        // h_next = h @ ws + agg @ wn + b  (agg read directly as f32)
        if (layer == 0) {
            sage_gemm<<<gemm_blocks, 256, 0, stream>>>(
                H[cur], wsT[0], (const void*)G32, wnT[0], 1, bb[0],
                H[cur ^ 1], (float*)nullptr, N, 1);
        } else if (layer == 1) {
            // pre-BN written in place over G32 (each wave owns its rows)
            sage_gemm<<<gemm_blocks, 256, 0, stream>>>(
                H[cur], wsT[1], (const void*)G32, wnT[1], 1, bb[1],
                (__bf16*)nullptr, G32, N, 2);
            zero_u32<<<1, 256, 0, stream>>>((unsigned*)stats, 64);
            bn_reduce<<<400, 128, 0, stream>>>(G32, sums, sumsq, N);
            bn_finalize<<<1, 128, 0, stream>>>(sums, sumsq, gamma, beta, scale, shift, N);
            bn_apply<<<elem_blocks, 256, 0, stream>>>(G32, scale, shift, H[cur ^ 1], N * 128);
        } else {
            sage_gemm<<<gemm_blocks, 256, 0, stream>>>(
                H[cur], wsT[2], (const void*)G32, wnT[2], 1, bb[2],
                H[cur ^ 1], (float*)nullptr, N, 2);
        }
        cur ^= 1;
    }
    sage_classifier<<<gemm_blocks, 256, 0, stream>>>(H[cur], wcT, bc, (float*)d_out, N);
    (void)ws_size; (void)out_size; (void)n_in;
}
